// TestNet_32298154066581
// MI455X (gfx1250) — matrix operand order repaired and checked by reading, not compiled, not run on hardware
//
#include <hip/hip_runtime.h>
#include <hip/hip_bf16.h>
#include <cstdint>
#include <cstddef>

typedef __attribute__((ext_vector_type(16))) _Float16 v16h;
typedef __attribute__((ext_vector_type(8)))  float    v8f;

constexpr int   B_   = 8;
constexpr int   N_   = 16384;
constexpr int   G_   = 64;
constexpr int   C_   = 384;
constexpr int   NPTS = B_ * N_;        // 131072
constexpr int   SEGS = B_ * G_;        // 512
constexpr float EPS_ = 1e-5f;

// ---------------- workspace layout (bytes) ----------------
constexpr size_t O_H    = 0;                         // h f16 [131072][256]
constexpr size_t O_REL  = O_H    + (size_t)NPTS*256*2;   // rel f32 [131072][3]
constexpr size_t O_SEG  = O_REL  + (size_t)NPTS*3*4;
constexpr size_t O_PERM = O_SEG  + (size_t)NPTS*4;
constexpr size_t O_SEGS = O_PERM + (size_t)NPTS*4;
constexpr size_t O_HIST = O_SEGS + (size_t)NPTS*4;
constexpr size_t O_OFFS = O_HIST + SEGS*4;
constexpr size_t O_FGU  = O_OFFS + SEGS*4;               // encoded fglob u32 [512][256]
constexpr size_t O_FG16 = O_FGU  + (size_t)SEGS*256*4;   // fglob f16 [512][256]
constexpr size_t O_P    = O_FG16 + (size_t)SEGS*256*2;   // P f32 [512][512]
constexpr size_t O_W2P  = O_P    + (size_t)SEGS*512*4;   // packed f16
constexpr size_t O_W3TP = O_W2P  + (size_t)128*256*2;
constexpr size_t O_W3BP = O_W3TP + (size_t)256*512*2;
constexpr size_t O_W4P  = O_W3BP + (size_t)256*512*2;
constexpr size_t O_S1   = O_W4P  + (size_t)512*384*2;
constexpr size_t O_T1   = O_S1 + 128*4;
constexpr size_t O_S2   = O_T1 + 128*4;
constexpr size_t O_T2   = O_S2 + 512*4;
constexpr size_t O_OUTU = O_T2 + 512*4;                  // encoded out u32 [512][384]

// phase2 dynamic LDS layout
constexpr unsigned SM_HBUF = 0;                       // f16 64*264
constexpr unsigned SM_TBUF = 64*264*2;                // f16 64*520  (=33792)
constexpr unsigned SM_S2   = SM_TBUF + 64*520*2;      // f32 512     (=100352)
constexpr unsigned SM_T2   = SM_S2 + 2048;
constexpr unsigned SM_SEGL = SM_T2 + 2048;            // u32 64
constexpr unsigned SMEM2   = SM_SEGL + 256;           // 104704 bytes

// ---------------- helpers ----------------
union Frag { v16h v; uint4 q[2]; };

__device__ __forceinline__ v8f wmma32(v16h a, v16h b, v8f c) {
  return __builtin_amdgcn_wmma_f32_16x16x32_f16(false, a, false, b, (short)0, c,
                                                false, false);
}

// A fragment (f16 16x32) from row-major LDS tile, ld in halves (mult of 8)
__device__ __forceinline__ v16h load_a_lds(const _Float16* base, int ld, int row0, int k0) {
  int lane = threadIdx.x & 31;
  int m = lane & 15, g = lane >> 4;
  const _Float16* p = base + (row0 + m) * ld + k0 + g * 8;
  Frag f;
  f.q[0] = *(const uint4*)p;
  f.q[1] = *(const uint4*)(p + 16);
  return f.v;
}

// B fragment from pre-packed weights: tile = kt*NT + nt, 512 halves/tile, 16/lane
__device__ __forceinline__ v16h load_b_packed(const _Float16* wp, int tile) {
  int lane = threadIdx.x & 31;
  const _Float16* p = wp + (size_t)tile * 512 + lane * 16;
  Frag f;
  f.q[0] = *(const uint4*)p;
  f.q[1] = *(const uint4*)(p + 8);
  return f.v;
}

__device__ __forceinline__ unsigned encf(float f) {
  unsigned u = __float_as_uint(f);
  return (u >> 31) ? ~u : (u | 0x80000000u);
}
__device__ __forceinline__ float decf(unsigned k) {
  unsigned u = (k >> 31) ? (k ^ 0x80000000u) : ~k;
  return __uint_as_float(u);
}

// ---------------- small prep kernels ----------------
__global__ __launch_bounds__(256) void k_fold(const float* g1, const float* be1,
                                              const float* m1, const float* v1,
                                              const float* g2, const float* be2,
                                              const float* m2, const float* v2,
                                              float* s1, float* t1, float* s2, float* t2) {
  int i = blockIdx.x * 256 + threadIdx.x;
  if (i < 512) { float s = g2[i] * rsqrtf(v2[i] + EPS_); s2[i] = s; t2[i] = be2[i] - m2[i] * s; }
  if (i < 128) { float s = g1[i] * rsqrtf(v1[i] + EPS_); s1[i] = s; t1[i] = be1[i] - m1[i] * s; }
}

__global__ __launch_bounds__(256) void k_pack(const float* W, int rowOff, int rowStride,
                                              int K, int N, _Float16* Wp) {
  int total = K * N;
  int NT = N >> 4;
  for (int idx = blockIdx.x * 256 + threadIdx.x; idx < total; idx += gridDim.x * 256) {
    int h = idx & 15, lane = (idx >> 4) & 31, tile = idx >> 9;
    int nt = tile % NT, kt = tile / NT;
    int k = kt * 32 + (h >> 3) * 16 + (lane >> 4) * 8 + (h & 7);
    int n = nt * 16 + (lane & 15);
    Wp[idx] = (_Float16)W[(size_t)(rowOff + k) * rowStride + n];
  }
}

// argmin over 64 centers + rel + histogram
__global__ __launch_bounds__(256) void k_assign(const float* data, const float* center,
                                                float* relG, unsigned* seg, unsigned* hist) {
  __shared__ float cs[192];
  int p = blockIdx.x * 256 + threadIdx.x;
  int b = p >> 14;  // N=16384
  if (threadIdx.x < 192) cs[threadIdx.x] = center[b * 192 + threadIdx.x];
  __syncthreads();
  float x = data[p * 3 + 0], y = data[p * 3 + 1], z = data[p * 3 + 2];
  float best = 3.4e38f; int bi = 0;
  for (int g = 0; g < 64; ++g) {
    float dx = x - cs[g * 3 + 0], dy = y - cs[g * 3 + 1], dz = z - cs[g * 3 + 2];
    float d2 = dx * dx + dy * dy + dz * dz;
    if (d2 < best) { best = d2; bi = g; }
  }
  relG[p * 3 + 0] = x - cs[bi * 3 + 0];
  relG[p * 3 + 1] = y - cs[bi * 3 + 1];
  relG[p * 3 + 2] = z - cs[bi * 3 + 2];
  unsigned s = (unsigned)(b * 64 + bi);
  seg[p] = s;
  atomicAdd(&hist[s], 1u);
}

__global__ __launch_bounds__(512) void k_scan(const unsigned* hist, unsigned* offs) {
  __shared__ unsigned s[512];
  int t = threadIdx.x;
  s[t] = hist[t];
  __syncthreads();
  for (int d = 1; d < 512; d <<= 1) {
    unsigned v = (t >= d) ? s[t - d] : 0u;
    __syncthreads();
    s[t] += v;
    __syncthreads();
  }
  offs[t] = s[t] - hist[t];  // exclusive
}

__global__ __launch_bounds__(256) void k_scatter(const unsigned* seg, unsigned* offs,
                                                 unsigned* perm, unsigned* segS) {
  int p = blockIdx.x * 256 + threadIdx.x;
  unsigned s = seg[p];
  unsigned j = atomicAdd(&offs[s], 1u);
  perm[j] = p;
  segS[j] = s;
}

// ---------------- phase 1: MM1 (VALU) + MM2 (WMMA) + fglob reduce ----------------
__global__ __launch_bounds__(256) void k_phase1(const float* relG, const unsigned* perm,
                                                const unsigned* segS, const float* W1,
                                                const float* b1, const float* s1g,
                                                const float* t1g, const _Float16* w2p,
                                                const float* b2, _Float16* hOut,
                                                unsigned* fgU) {
  __shared__ _Float16 xbuf[64 * 136];   // [64 pts][K=128], ld=136
  __shared__ _Float16 hb[64 * 264];     // [64 pts][256],   ld=264
  __shared__ float relb[64][3];
  __shared__ unsigned segsL[64];
  __shared__ float s1[128], t1[128];
  int tid = threadIdx.x;
  int j0 = blockIdx.x * 64;
  if (tid < 64) {
    unsigned pp = perm[j0 + tid];
    relb[tid][0] = relG[pp * 3 + 0];
    relb[tid][1] = relG[pp * 3 + 1];
    relb[tid][2] = relG[pp * 3 + 2];
    segsL[tid] = segS[j0 + tid];
  }
  if (tid < 128) { s1[tid] = s1g[tid]; t1[tid] = t1g[tid]; }
  __syncthreads();
  // MM1: relu(bn(rel @ W1 + b1)) -> xbuf f16
  for (int i = tid; i < 64 * 128; i += 256) {
    int r = i >> 7, f = i & 127;
    float x = relb[r][0] * W1[f] + relb[r][1] * W1[128 + f] + relb[r][2] * W1[256 + f] + b1[f];
    x = fmaxf(x * s1[f] + t1[f], 0.f);
    xbuf[r * 136 + f] = (_Float16)x;
  }
  __syncthreads();
  // MM2 (WMMA): h = x @ W2 + b2
  int w = tid >> 5, lane = tid & 31;
  int mt = w & 3, nh = w >> 2;
  int n = lane & 15, g = lane >> 4;
  for (int nt2 = 0; nt2 < 8; ++nt2) {
    int nt = nh * 8 + nt2;
    v8f c = {0.f, 0.f, 0.f, 0.f, 0.f, 0.f, 0.f, 0.f};
    for (int kt = 0; kt < 4; ++kt) {
      v16h a = load_a_lds(xbuf, 136, mt * 16, kt * 32);
      v16h b = load_b_packed(w2p, kt * 16 + nt);
      c = wmma32(a, b, c);
    }
    int col = nt * 16 + n;
    float bb = b2[col];
#pragma unroll
    for (int r = 0; r < 8; ++r) {
      int row = mt * 16 + r + 8 * g;
      hb[row * 264 + col] = (_Float16)(c[r] + bb);
    }
  }
  __syncthreads();
  // coalesced store of h rows (sorted order)
  {
    int row = tid >> 2, qi = tid & 3;
    const uint4* src = (const uint4*)(hb + row * 264 + qi * 64);
    uint4* dst = (uint4*)(hOut + (size_t)(j0 + row) * 256 + qi * 64);
#pragma unroll
    for (int i = 0; i < 8; ++i) dst[i] = src[i];
  }
  // per-block segment-max reduce, then ~1 atomic per (distinct seg, col)
  {
    int c = tid;  // 256 threads == 256 cols
    unsigned cur = 0, s = segsL[0];
    for (int r = 0; r < 64; ++r) {
      unsigned sr = segsL[r];
      if (sr != s) { atomicMax(&fgU[s * 256 + c], cur); cur = 0; s = sr; }
      unsigned k = encf((float)hb[r * 264 + c]);
      cur = (k > cur) ? k : cur;
    }
    atomicMax(&fgU[s * 256 + c], cur);
  }
}

__global__ __launch_bounds__(256) void k_fgdec(const unsigned* fgU, _Float16* fg16) {
  int i = blockIdx.x * 256 + threadIdx.x;  // 131072
  fg16[i] = (_Float16)decf(fgU[i]);
}

// ---------------- P = fglob @ W3[0:256] + b3  (512x512) ----------------
__global__ __launch_bounds__(256) void k_pgemm(const _Float16* fg16, const _Float16* w3tp,
                                               const float* b3, float* P) {
  int w = threadIdx.x >> 5, lane = threadIdx.x & 31;
  int id = blockIdx.x * 8 + w;  // 1024 tiles
  int mt = id >> 5, nt = id & 31;
  int n = lane & 15, g = lane >> 4;
  int m = lane & 15;
  v8f c = {0.f, 0.f, 0.f, 0.f, 0.f, 0.f, 0.f, 0.f};
  for (int kt = 0; kt < 8; ++kt) {
    const _Float16* p = fg16 + (size_t)(mt * 16 + m) * 256 + kt * 32 + g * 8;
    Frag fa;
    fa.q[0] = *(const uint4*)p;
    fa.q[1] = *(const uint4*)(p + 16);
    v16h b = load_b_packed(w3tp, kt * 32 + nt);
    c = wmma32(fa.v, b, c);
  }
  int col = nt * 16 + n;
  float bb = b3[col];
#pragma unroll
  for (int r = 0; r < 8; ++r)
    P[(size_t)(mt * 16 + r + 8 * g) * 512 + col] = c[r] + bb;
}

// ---------------- phase 2: GEMM3 + bn/relu + GEMM4 + out reduce ----------------
__global__ __launch_bounds__(256) void k_phase2(const _Float16* hG, const unsigned* segS,
                                                const float* P, const _Float16* w3bp,
                                                const _Float16* w4p, const float* s2g,
                                                const float* t2g, unsigned* outU) {
  extern __shared__ char smem[];
  _Float16* hbuf = (_Float16*)(smem + SM_HBUF);   // 64 x 264
  _Float16* tbuf = (_Float16*)(smem + SM_TBUF);   // 64 x 520
  float* s2 = (float*)(smem + SM_S2);
  float* t2 = (float*)(smem + SM_T2);
  unsigned* segsL = (unsigned*)(smem + SM_SEGL);
  unsigned* h2enc = (unsigned*)smem;              // overlay 64 x 392 u32

  int tid = threadIdx.x;
  int j0 = blockIdx.x * 64;
  if (tid < 64) segsL[tid] = segS[j0 + tid];
  for (int i = tid; i < 512; i += 256) { s2[i] = s2g[i]; t2[i] = t2g[i]; }
  {
    int row = tid >> 2, qi = tid & 3;
    const uint4* src = (const uint4*)(hG + (size_t)(j0 + row) * 256 + qi * 64);
    uint4* dst = (uint4*)(hbuf + row * 264 + qi * 64);
#pragma unroll
    for (int i = 0; i < 8; ++i) dst[i] = src[i];
  }
  __syncthreads();
  int w = tid >> 5, lane = tid & 31;
  int n = lane & 15, g = lane >> 4;
  // GEMM3: t = relu(bn(P[seg] + h @ W3bot))
  {
    int mt = w & 3, nh = w >> 2;
    for (int nt2 = 0; nt2 < 16; ++nt2) {
      int nt = nh * 16 + nt2;
      int col = nt * 16 + n;
      v8f c;
#pragma unroll
      for (int r = 0; r < 8; ++r)
        c[r] = P[(size_t)segsL[mt * 16 + r + 8 * g] * 512 + col];
      for (int kt = 0; kt < 8; ++kt) {
        v16h a = load_a_lds(hbuf, 264, mt * 16, kt * 32);
        v16h b = load_b_packed(w3bp, kt * 32 + nt);
        c = wmma32(a, b, c);
      }
      float ss = s2[col], tt = t2[col];
#pragma unroll
      for (int r = 0; r < 8; ++r) {
        int row = mt * 16 + r + 8 * g;
        tbuf[row * 520 + col] = (_Float16)fmaxf(c[r] * ss + tt, 0.f);
      }
    }
  }
  __syncthreads();
  // GEMM4: h2 = t @ W4 (b4 added in decode; max commutes with +const)
  {
    int mt = w & 3, nq = w >> 2;
    v8f acc[12];
#pragma unroll
    for (int i = 0; i < 12; ++i) acc[i] = (v8f){0.f, 0.f, 0.f, 0.f, 0.f, 0.f, 0.f, 0.f};
    for (int kt = 0; kt < 16; ++kt) {
      v16h a = load_a_lds(tbuf, 520, mt * 16, kt * 32);
#pragma unroll
      for (int nt2 = 0; nt2 < 12; ++nt2) {
        int nt = nq * 12 + nt2;
        v16h b = load_b_packed(w4p, kt * 24 + nt);
        acc[nt2] = wmma32(a, b, acc[nt2]);
      }
    }
    __syncthreads();  // everyone done with hbuf/tbuf before overlay
#pragma unroll
    for (int nt2 = 0; nt2 < 12; ++nt2) {
      int col = (nq * 12 + nt2) * 16 + n;
#pragma unroll
      for (int r = 0; r < 8; ++r) {
        int row = mt * 16 + r + 8 * g;
        h2enc[row * 392 + col] = encf(acc[nt2][r]);
      }
    }
  }
  __syncthreads();
  // per-block reduce over sorted segments, then sparse atomics
  for (int c = tid; c < 384; c += 256) {
    unsigned cur = 0, s = segsL[0];
    for (int r = 0; r < 64; ++r) {
      unsigned sr = segsL[r];
      if (sr != s) { atomicMax(&outU[s * 384 + c], cur); cur = 0; s = sr; }
      unsigned k = h2enc[r * 392 + c];
      cur = (k > cur) ? k : cur;
    }
    atomicMax(&outU[s * 384 + c], cur);
  }
}

__global__ __launch_bounds__(256) void k_outdec(const unsigned* outU, const float* b4,
                                                float* out) {
  int i = blockIdx.x * 256 + threadIdx.x;  // 196608
  int c = i % 384;
  out[i] = decf(outU[i]) + b4[c];
}

// ---------------- launcher ----------------
extern "C" void kernel_launch(void* const* d_in, const int* in_sizes, int n_in,
                              void* d_out, int out_size, void* d_ws, size_t ws_size,
                              hipStream_t stream) {
  const float* data   = (const float*)d_in[0];
  const float* center = (const float*)d_in[1];
  const float* W1 = (const float*)d_in[2];
  const float* b1 = (const float*)d_in[3];
  const float* g1 = (const float*)d_in[4];
  const float* be1 = (const float*)d_in[5];
  const float* m1 = (const float*)d_in[6];
  const float* v1 = (const float*)d_in[7];
  const float* W2 = (const float*)d_in[8];
  const float* b2 = (const float*)d_in[9];
  const float* W3 = (const float*)d_in[10];
  const float* b3 = (const float*)d_in[11];
  const float* g2 = (const float*)d_in[12];
  const float* be2 = (const float*)d_in[13];
  const float* m2 = (const float*)d_in[14];
  const float* v2 = (const float*)d_in[15];
  const float* W4 = (const float*)d_in[16];
  const float* b4 = (const float*)d_in[17];

  char* ws = (char*)d_ws;
  _Float16* hF   = (_Float16*)(ws + O_H);
  float*    relG = (float*)(ws + O_REL);
  unsigned* seg  = (unsigned*)(ws + O_SEG);
  unsigned* perm = (unsigned*)(ws + O_PERM);
  unsigned* segS = (unsigned*)(ws + O_SEGS);
  unsigned* hist = (unsigned*)(ws + O_HIST);
  unsigned* offs = (unsigned*)(ws + O_OFFS);
  unsigned* fgU  = (unsigned*)(ws + O_FGU);
  _Float16* fg16 = (_Float16*)(ws + O_FG16);
  float*    P    = (float*)(ws + O_P);
  _Float16* w2p  = (_Float16*)(ws + O_W2P);
  _Float16* w3tp = (_Float16*)(ws + O_W3TP);
  _Float16* w3bp = (_Float16*)(ws + O_W3BP);
  _Float16* w4p  = (_Float16*)(ws + O_W4P);
  float* s1 = (float*)(ws + O_S1);
  float* t1 = (float*)(ws + O_T1);
  float* s2 = (float*)(ws + O_S2);
  float* t2 = (float*)(ws + O_T2);
  unsigned* outU = (unsigned*)(ws + O_OUTU);

  hipMemsetAsync(hist, 0, SEGS * 4, stream);
  hipMemsetAsync(fgU, 0, (size_t)SEGS * 256 * 4, stream);
  hipMemsetAsync(outU, 0, (size_t)SEGS * 384 * 4, stream);

  k_fold<<<2, 256, 0, stream>>>(g1, be1, m1, v1, g2, be2, m2, v2, s1, t1, s2, t2);
  k_pack<<<128, 256, 0, stream>>>(W2, 0, 256, 128, 256, w2p);
  k_pack<<<512, 256, 0, stream>>>(W3, 0, 512, 256, 512, w3tp);
  k_pack<<<512, 256, 0, stream>>>(W3, 256, 512, 256, 512, w3bp);
  k_pack<<<768, 256, 0, stream>>>(W4, 0, 384, 512, 384, w4p);

  k_assign<<<NPTS / 256, 256, 0, stream>>>(data, center, relG, seg, hist);
  k_scan<<<1, 512, 0, stream>>>(hist, offs);
  k_scatter<<<NPTS / 256, 256, 0, stream>>>(seg, offs, perm, segS);

  k_phase1<<<NPTS / 64, 256, 0, stream>>>(relG, perm, segS, W1, b1, s1, t1, w2p, b2,
                                          hF, fgU);
  k_fgdec<<<(SEGS * 256) / 256, 256, 0, stream>>>(fgU, fg16);
  k_pgemm<<<128, 256, 0, stream>>>(fg16, w3tp, b3, P);

  hipFuncSetAttribute((const void*)k_phase2,
                      hipFuncAttributeMaxDynamicSharedMemorySize, (int)SMEM2);
  k_phase2<<<NPTS / 64, 256, SMEM2, stream>>>(hF, segS, P, w3bp, w4p, s2, t2, outU);

  k_outdec<<<(SEGS * 384) / 256, 256, 0, stream>>>(outU, b4, (float*)d_out);
}
